// VQSoftAttention_22368189677591
// MI455X (gfx1250) — matrix operand order repaired and checked by reading, not compiled, not run on hardware
//
#include <hip/hip_runtime.h>
#include <hip/hip_bf16.h>
#include <math.h>

// ---------------- problem constants ----------------
constexpr int BATCH = 16384;
constexpr int DIM   = 512;
constexpr int KCB   = 1024;
constexpr float ALPHA = 10.0f;
constexpr float BETA  = 0.25f;

// output layout (floats), tuple concatenated flat:
constexpr size_t Q_OFF    = 0;                       // quantized   [B*D]
constexpr size_t VQ_OFF   = 8388608;                 // vq_loss     [1]
constexpr size_t ENT_OFF  = 8388609;                 // entropy_vq  [1]
constexpr size_t ARGM_OFF = 8388610;                 // argm        [B]
constexpr size_t HQ_OFF   = 8404994;                 // hard_quant  [B*D]
constexpr size_t SA_OFF   = 16793602;                // soft_assign [B*K]
constexpr size_t CM_OFF   = 33570818;                // cluster_metric [1]

// workspace layout (bytes)
constexpr size_t WS_ACC    = 0;          // float[1026]: [0]=sse [1]=rowmax_sum [2..1025]=avg sums
constexpr size_t WS_INVN   = 8192;       // float[16384]
constexpr size_t WS_LN16   = 73728;      // _Float16[B*D]   (16 MB)
constexpr size_t WS_CB16   = 16850944;   // _Float16[K*D]   (1 MB)   normalized codebook
constexpr size_t WS_EMBT16 = 17899520;   // _Float16[D*K]   (1 MB)   transposed raw codebook

// ---------------- WMMA vector types ----------------
typedef __attribute__((ext_vector_type(16))) _Float16 v16h;
typedef __attribute__((ext_vector_type(8)))  _Float16 v8h;
typedef __attribute__((ext_vector_type(8)))  float    v8f;
typedef __attribute__((ext_vector_type(4)))  int      v4i;

union ABfrag { v16h v; v8h h[2]; };

// CDNA5 async global->LDS path (builtin probe-confirmed present; typed AS ptrs)
#if defined(__has_builtin)
# if __has_builtin(__builtin_amdgcn_global_load_async_to_lds_b128)
#  define HAVE_ASYNC_LDS 1
# endif
#endif
#ifndef HAVE_ASYNC_LDS
# define HAVE_ASYNC_LDS 0
#endif

typedef __attribute__((address_space(1))) v4i* gptr_v4i;
typedef __attribute__((address_space(3))) v4i* lptr_v4i;

__device__ __forceinline__ void wait_async_lds() {
#if HAVE_ASYNC_LDS
# if defined(__has_builtin) && __has_builtin(__builtin_amdgcn_s_wait_asynccnt)
  __builtin_amdgcn_s_wait_asynccnt(0);
# else
  asm volatile("s_wait_asynccnt 0x0" ::: "memory");
# endif
#endif
}

__device__ __forceinline__ float wave_sum(float x) {
  #pragma unroll
  for (int m = 16; m >= 1; m >>= 1) x += __shfl_xor(x, m, 32);
  return x;
}

// ---------------- kernel 0: zero accumulators ----------------
__global__ void k_zero(float* acc) {
  for (int i = threadIdx.x; i < 1026; i += blockDim.x) acc[i] = 0.0f;
}

// ---------------- kernel 1: codebook prep ----------------
__global__ __launch_bounds__(256) void k_prep_emb(const float* __restrict__ emb,
                                                  _Float16* __restrict__ cb16,
                                                  _Float16* __restrict__ embT16) {
  const int k = blockIdx.x;
  const int tid = threadIdx.x;
  const float* row = emb + (size_t)k * DIM;
  float v0 = row[tid];
  float v1 = row[tid + 256];
  float ss = wave_sum(v0 * v0 + v1 * v1);
  __shared__ float lred[8];
  const int w = tid >> 5, l = tid & 31;
  if (l == 0) lred[w] = ss;
  __syncthreads();
  float tot = 0.f;
  #pragma unroll
  for (int i = 0; i < 8; ++i) tot += lred[i];
  const float scale = 1.0f / fmaxf(sqrtf(tot), 1e-12f);
  cb16[(size_t)k * DIM + tid]        = (_Float16)(v0 * scale);
  cb16[(size_t)k * DIM + tid + 256]  = (_Float16)(v1 * scale);
  embT16[(size_t)tid * KCB + k]         = (_Float16)v0;
  embT16[(size_t)(tid + 256) * KCB + k] = (_Float16)v1;
}

// ---------------- kernel 2: latent prep ----------------
__global__ __launch_bounds__(256) void k_prep_lat(const float* __restrict__ lat,
                                                  _Float16* __restrict__ ln16,
                                                  float* __restrict__ invn) {
  const int w = threadIdx.x >> 5, l = threadIdx.x & 31;
  const int row = blockIdx.x * 8 + w;
  const float* src = lat + (size_t)row * DIM;
  float vals[16];
  float ss = 0.f;
  #pragma unroll
  for (int j = 0; j < 16; ++j) { vals[j] = src[l + 32 * j]; ss += vals[j] * vals[j]; }
  ss = wave_sum(ss);
  const float scale = 1.0f / fmaxf(sqrtf(ss), 1e-12f);
  _Float16* dst = ln16 + (size_t)row * DIM;
  #pragma unroll
  for (int j = 0; j < 16; ++j) dst[l + 32 * j] = (_Float16)(vals[j] * scale);
  if (l == 0) invn[row] = scale;
}

// ---------------- kernel 3: fused GEMM1 -> softmax -> GEMM2 ----------------
// 32 latent rows per block (halves cb/embT L2 streams vs 16-row tiles).
// LDS: att[16][1024] f32 (reused per 16-row half) + sa16[32][1024] f16 + ln tile.
__global__ __launch_bounds__(256) void k_fused(const float* __restrict__ latents,
                                               const float* __restrict__ emb,
                                               const _Float16* __restrict__ ln16,
                                               const _Float16* __restrict__ cb16,
                                               const _Float16* __restrict__ embT16,
                                               const float* __restrict__ invn,
                                               float* __restrict__ acc_ws,
                                               float* __restrict__ out) {
  extern __shared__ char smem[];
  float*    att   = (float*)smem;                // [16][1024] = 64 KB
  _Float16* sa16  = (_Float16*)(smem + 65536);   // [32][1024] = 64 KB
  _Float16* lnlds = (_Float16*)(smem + 131072);  // [32][512]  = 32 KB

  const int tid = threadIdx.x;
  const int w = tid >> 5, l = tid & 31;
  const int lmod = l & 15;
  const int lhi  = l >> 4;            // 0 for lanes 0-15, 1 for lanes 16-31
  const int row0 = blockIdx.x * 32;

  // ---- stage 32x512 f16 latent tile into LDS (async if available) ----
  {
    const _Float16* src = ln16 + (size_t)row0 * DIM;   // 32 KB contiguous
    #pragma unroll
    for (int j = 0; j < 8; ++j) {
      const int idx = (tid + 256 * j) * 8;             // halves, 16B chunks
#if HAVE_ASYNC_LDS
      __builtin_amdgcn_global_load_async_to_lds_b128(
          (gptr_v4i)(src + idx), (lptr_v4i)(lnlds + idx), 0, 0);
#else
      *(v8h*)(lnlds + idx) = *(const v8h*)(src + idx);
#endif
    }
    wait_async_lds();
    __syncthreads();
  }

  // ===== GEMM1: att[32 x 1024] = ln * cb^T ; wave w owns col tiles 8w..8w+7 =====
  v8f acc[16];   // [mi*8 + j], mi = row half (0: rows 0-15, 1: rows 16-31)
  #pragma unroll
  for (int j = 0; j < 16; ++j) acc[j] = (v8f){0.f,0.f,0.f,0.f,0.f,0.f,0.f,0.f};

  const _Float16* arow0 = lnlds + (size_t)lmod * DIM;
  const _Float16* arow1 = lnlds + (size_t)(16 + lmod) * DIM;
  for (int ds = 0; ds < 16; ++ds) {
    const int d0 = ds * 32;
    ABfrag a0, a1;
    a0.h[0] = *(const v8h*)(arow0 + d0 + lhi * 8);
    a0.h[1] = *(const v8h*)(arow0 + d0 + 16 + lhi * 8);
    a1.h[0] = *(const v8h*)(arow1 + d0 + lhi * 8);
    a1.h[1] = *(const v8h*)(arow1 + d0 + 16 + lhi * 8);
    #pragma unroll
    for (int j = 0; j < 8; ++j) {
      const int kcol = (w * 8 + j) * 16 + lmod;
      const _Float16* brow = cb16 + (size_t)kcol * DIM + d0 + lhi * 8;
      __builtin_prefetch(brow + 32, 0, 1);   // next K-step, per-lane row
      ABfrag b;
      b.h[0] = *(const v8h*)(brow);
      b.h[1] = *(const v8h*)(brow + 16);
      acc[j] = __builtin_amdgcn_wmma_f32_16x16x32_f16(
          false, a0.v, false, b.v, (short)0, acc[j], false, false);
      acc[8 + j] = __builtin_amdgcn_wmma_f32_16x16x32_f16(
          false, a1.v, false, b.v, (short)0, acc[8 + j], false, false);
    }
  }

  // ===== per 16-row half: spill -> softmax/argmax -> hard_quantized =====
  for (int h = 0; h < 2; ++h) {
    if (h) __syncthreads();          // protect att overwrite vs prior reads
    #pragma unroll
    for (int j = 0; j < 8; ++j) {
      const int n = (w * 8 + j) * 16 + lmod;
      #pragma unroll
      for (int r = 0; r < 8; ++r)
        att[(r + lhi * 8) * 1024 + n] = acc[h * 8 + j][r];
    }
    __syncthreads();

    for (int rr = 0; rr < 2; ++rr) {
      const int m = w * 2 + rr;              // local row in this half
      const float* arow2 = att + m * 1024;
      float bv = -3.0e38f; int bi = 0;
      for (int j = 0; j < 32; ++j) {
        const int c = l + 32 * j;
        const float v = arow2[c];
        if (v > bv) { bv = v; bi = c; }      // strict > keeps first index
      }
      #pragma unroll
      for (int mask = 16; mask >= 1; mask >>= 1) {
        const float ov = __shfl_xor(bv, mask, 32);
        const int   oi = __shfl_xor(bi, mask, 32);
        if (ov > bv || (ov == bv && oi < bi)) { bv = ov; bi = oi; }
      }
      float s = 0.f;
      for (int j = 0; j < 32; ++j)
        s += __expf(ALPHA * (arow2[l + 32 * j] - bv));
      s = wave_sum(s);
      const float inv_s = 1.0f / s;
      const int grow = row0 + h * 16 + m;
      float* sarow_out = out + SA_OFF + (size_t)grow * KCB;
      for (int j = 0; j < 32; ++j) {
        const int c = l + 32 * j;
        const float p = __expf(ALPHA * (arow2[c] - bv)) * inv_s;
        sarow_out[c] = p;
        sa16[(h * 16 + m) * 1024 + c] = (_Float16)p;
      }
      if (l == 0) {
        out[ARGM_OFF + grow] = (float)bi;
        atomicAdd(&acc_ws[1], bv);           // cluster_metric numerator
      }
      const float2* src = (const float2*)(emb + (size_t)bi * DIM);
      float2* dst = (float2*)(out + HQ_OFF + (size_t)grow * DIM);
      #pragma unroll
      for (int j = 0; j < 8; ++j) dst[l + 32 * j] = src[l + 32 * j];
    }
  }
  __syncthreads();

  // ===== avg_probs partial column sums over 32 rows =====
  #pragma unroll
  for (int q = 0; q < 4; ++q) {
    const int col = tid + 256 * q;
    float cs = 0.f;
    #pragma unroll
    for (int r = 0; r < 32; ++r) cs += (float)sa16[r * 1024 + col];
    atomicAdd(&acc_ws[2 + col], cs);
  }

  // ===== GEMM2: quantized[32 x 512] = sa16 * emb ; wave w owns d tiles 4w..4w+3 =====
  v8f acc2[8];   // [mi*4 + j]
  #pragma unroll
  for (int j = 0; j < 8; ++j) acc2[j] = (v8f){0.f,0.f,0.f,0.f,0.f,0.f,0.f,0.f};

  for (int kk = 0; kk < 32; ++kk) {
    const int k0 = kk * 32;
    ABfrag a0, a1;
    const _Float16* sar0 = sa16 + (size_t)lmod * 1024 + k0 + lhi * 8;
    const _Float16* sar1 = sa16 + (size_t)(16 + lmod) * 1024 + k0 + lhi * 8;
    a0.h[0] = *(const v8h*)(sar0);
    a0.h[1] = *(const v8h*)(sar0 + 16);
    a1.h[0] = *(const v8h*)(sar1);
    a1.h[1] = *(const v8h*)(sar1 + 16);
    #pragma unroll
    for (int j = 0; j < 4; ++j) {
      const int dcol = (w * 4 + j) * 16 + lmod;
      const _Float16* brow = embT16 + (size_t)dcol * KCB + k0 + lhi * 8;
      __builtin_prefetch(brow + 32, 0, 1);
      ABfrag b;
      b.h[0] = *(const v8h*)(brow);
      b.h[1] = *(const v8h*)(brow + 16);
      acc2[j] = __builtin_amdgcn_wmma_f32_16x16x32_f16(
          false, a0.v, false, b.v, (short)0, acc2[j], false, false);
      acc2[4 + j] = __builtin_amdgcn_wmma_f32_16x16x32_f16(
          false, a1.v, false, b.v, (short)0, acc2[4 + j], false, false);
    }
  }

  // epilogue: store quantized + fused MSE vs ln (latents * inv_norm)
  float sse = 0.f;
  #pragma unroll
  for (int mi = 0; mi < 2; ++mi) {
    #pragma unroll
    for (int j = 0; j < 4; ++j) {
      const int d = (w * 4 + j) * 16 + lmod;
      #pragma unroll
      for (int r = 0; r < 8; ++r) {
        const int grow = row0 + mi * 16 + r + lhi * 8;
        const float q = acc2[mi * 4 + j][r];
        out[Q_OFF + (size_t)grow * DIM + d] = q;
        const float lf = latents[(size_t)grow * DIM + d] * invn[grow];
        const float df = q - lf;
        sse += df * df;
      }
    }
  }
  sse = wave_sum(sse);
  if (l == 0) atomicAdd(&acc_ws[0], sse);
}

// ---------------- kernel 4: finalize scalars ----------------
__global__ __launch_bounds__(1024) void k_final(const float* __restrict__ acc_ws,
                                                float* __restrict__ out) {
  __shared__ float red[32];
  const int tid = threadIdx.x;
  const float p = acc_ws[2 + tid] / (float)BATCH;
  float e = -p * logf(p + 1e-10f);
  e = wave_sum(e);
  const int w = tid >> 5, l = tid & 31;
  if (l == 0) red[w] = e;
  __syncthreads();
  if (tid == 0) {
    float tot = 0.f;
    #pragma unroll
    for (int i = 0; i < 32; ++i) tot += red[i];
    out[ENT_OFF] = tot;
    out[VQ_OFF]  = (1.0f + BETA) * acc_ws[0] / ((float)BATCH * (float)DIM);
    out[CM_OFF]  = acc_ws[1] / (float)BATCH;
  }
}

// ---------------- host launcher ----------------
extern "C" void kernel_launch(void* const* d_in, const int* in_sizes, int n_in,
                              void* d_out, int out_size, void* d_ws, size_t ws_size,
                              hipStream_t stream) {
  (void)in_sizes; (void)n_in; (void)out_size; (void)ws_size;
  const float* latents = (const float*)d_in[0];
  const float* emb     = (const float*)d_in[1];
  float* out = (float*)d_out;
  char*  ws  = (char*)d_ws;

  float*    acc    = (float*)(ws + WS_ACC);
  float*    invn   = (float*)(ws + WS_INVN);
  _Float16* ln16   = (_Float16*)(ws + WS_LN16);
  _Float16* cb16   = (_Float16*)(ws + WS_CB16);
  _Float16* embT16 = (_Float16*)(ws + WS_EMBT16);

  k_zero<<<1, 256, 0, stream>>>(acc);
  k_prep_emb<<<KCB, 256, 0, stream>>>(emb, cb16, embT16);
  k_prep_lat<<<BATCH / 8, 256, 0, stream>>>(latents, ln16, invn);

  const size_t lds_bytes = 65536 + 65536 + 32768;   // att + sa16 + ln tile = 160 KB
  k_fused<<<BATCH / 32, 256, lds_bytes, stream>>>(latents, emb, ln16, cb16,
                                                  embT16, invn, acc, out);
  k_final<<<1, 1024, 0, stream>>>(acc, out);
}
